// MultiHeadAttention_8074538516581
// MI455X (gfx1250) — matrix operand order repaired and checked by reading, compiled, not run on hardware
//
#include <hip/hip_runtime.h>

// ---------------------------------------------------------------------------
// MI455X / gfx1250 fused multi-head-attention block.
// Heavy math on v_wmma_f32_16x16x32_bf16; attention K-tile staging via the
// Tensor Data Mover (tensor_load_to_lds + s_wait_tensorcnt) when available,
// next-tile global_prefetch, fixed-max softmax (valid because the reference
// clamps logits to [-30,30] before the row max, so exp(s-m) <= 1 always).
// ---------------------------------------------------------------------------

typedef __attribute__((ext_vector_type(16))) __bf16 v16bf;
typedef __attribute__((ext_vector_type(8)))  __bf16 v8bf;
typedef __attribute__((ext_vector_type(8)))  float  v8f;
typedef __attribute__((ext_vector_type(4)))  unsigned int u32x4;
typedef __attribute__((ext_vector_type(8)))  int i32x8;
typedef __attribute__((ext_vector_type(4)))  int i32x4;

#define NHEADS 12
#define HDIM   64
#define TLEN   1024
#define BATCH  4
#define EMB    768
#define NTOK   (BATCH * TLEN)        // 4096
#define QKVN   (3 * NHEADS * HDIM)   // 2304

#if __has_builtin(__builtin_amdgcn_tensor_load_to_lds) && \
    __has_builtin(__builtin_amdgcn_s_wait_tensorcnt)
#define MHA_HAS_TDM 1
#if __has_include(<hip/amd_detail/amd_gfx1250_TDM.h>)
#define MHA_TDM_6ARG 1   // therock-10.0 headers => 6-arg builtin
#endif
#endif

static __device__ inline v8f wmma_bf16(v16bf a, v16bf b, v8f c) {
  // 8 args: (neg_a, A, neg_b, B, c_mod, C, reuse_a, reuse_b)
  return __builtin_amdgcn_wmma_f32_16x16x32_bf16(
      false, a, false, b, (short)0, c, false, false);
}

static __device__ inline v16bf zero16bf() {
  v16bf z;
#pragma unroll
  for (int i = 0; i < 16; ++i) z[i] = (__bf16)0.0f;
  return z;
}

// A-matrix (16x32 bf16) per-lane load from a row-major row pointer.
// Lane L holds row M=L&15; elements [0:16) = K[2*base..2*base+16),
// base = (lane<16) ? 0 : 8 (contiguous order, matches B loads).
static __device__ inline v16bf load_a16(const __bf16* __restrict__ rowp, int base) {
  v8bf lo = *(const v8bf*)(rowp + 2 * base);
  v8bf hi = *(const v8bf*)(rowp + 2 * base + 8);
  v16bf a;
#pragma unroll
  for (int i = 0; i < 8; ++i) { a[i] = lo[i]; a[8 + i] = hi[i]; }
  return a;
}

#ifdef MHA_HAS_TDM
// Issue a TDM load of a [rows x 64] bf16 tile (row-major, stride 64 elems)
// from global memory into LDS at byte offset lds_off. ISA ch.8 D# layout.
static __device__ inline void tdm_load_tile_64w(unsigned long long gaddr,
                                                unsigned lds_off, int rows) {
  u32x4 g0 = {1u,                                     // count=1, user mode
              lds_off,                                // lds_addr (bytes)
              (unsigned)(gaddr & 0xffffffffull),      // global_addr[31:0]
              (unsigned)((gaddr >> 32) & 0x01ffffffull) | (2u << 30)}; // +type=2
  i32x8 g1 = {(int)(1u << 16),                        // data_size=1 (2 bytes)
              (int)(64u << 16),                       // tensor_dim0 = 64
              (int)((unsigned)TLEN << 16),            // tensor_dim1 = 1024
              (int)(64u << 16),                       // tile_dim0 = 64
              rows,                                   // tile_dim1 = rows
              64,                                     // tensor_dim0_stride = 64
              0, 0};
  i32x4 z4 = {0, 0, 0, 0};
#ifdef MHA_TDM_6ARG
  i32x8 z8 = {0, 0, 0, 0, 0, 0, 0, 0};
  __builtin_amdgcn_tensor_load_to_lds(g0, g1, z4, z4, z8, 0);
#else
  __builtin_amdgcn_tensor_load_to_lds(g0, g1, z4, z4, 0);
#endif
}

static __device__ inline unsigned lds_byte_offset(const void* p) {
  typedef __attribute__((address_space(3))) char lds_char;
  return (unsigned)(unsigned long long)(lds_char*)p;
}
#endif

// ---------------------------------------------------------------------------
// Kernel 1: x' = x + struct_embed[struct_id(idx)], stored bf16.
// ---------------------------------------------------------------------------
__global__ void mha_add_struct_kernel(const float* __restrict__ x,
                                      const int* __restrict__ idx,
                                      const float* __restrict__ se,
                                      __bf16* __restrict__ xb, int n) {
  int i = blockIdx.x * blockDim.x + threadIdx.x;
  if (i >= n) return;
  int tok = i / EMB;
  int e   = i - tok * EMB;
  int id  = idx[tok];
  int sid = (id == 1) ? 1 : (id == 2) ? 2 : (id == 3) ? 3 : 0;
  xb[i] = (__bf16)(x[i] + se[sid * EMB + e]);
}

// ---------------------------------------------------------------------------
// Kernel 2: fp32 -> bf16 convert (weights).
// ---------------------------------------------------------------------------
__global__ void mha_cvt_bf16_kernel(const float* __restrict__ src,
                                    __bf16* __restrict__ dst, int n) {
  int i = blockIdx.x * blockDim.x + threadIdx.x;
  if (i < n) dst[i] = (__bf16)src[i];
}

// ---------------------------------------------------------------------------
// Kernel 3: QKV GEMM. Each wave computes a 16x64 strip (4 N-tiles share one
// A fragment per K-step -> 4 WMMAs per A load). 4 waves/block on distinct M.
// Results scattered bf16 into q/k/v [B,H,T,D].
// ---------------------------------------------------------------------------
__global__ __launch_bounds__(128) void mha_qkv_gemm_kernel(
    const __bf16* __restrict__ xb, const __bf16* __restrict__ wq,
    __bf16* __restrict__ q, __bf16* __restrict__ k, __bf16* __restrict__ v) {
  int lane = threadIdx.x & 31;
  int wave = threadIdx.x >> 5;
  int n0 = blockIdx.x * 64;
  int m0 = (blockIdx.y * 4 + wave) * 16;

  int arow  = m0 + (lane & 15);
  int abase = (lane < 16) ? 0 : 8;
  int bk0   = (lane < 16) ? 0 : 16;   // B: lane holds col N, K rows k0..k0+15

  const __bf16* ap = xb + (size_t)arow * EMB;
  const __bf16* bp[4];
#pragma unroll
  for (int j = 0; j < 4; ++j)
    bp[j] = wq + (size_t)(n0 + j * 16 + (lane & 15)) * EMB + bk0;

  v8f c[4];
#pragma unroll
  for (int j = 0; j < 4; ++j) { v8f z = {}; c[j] = z; }

  for (int kk = 0; kk < EMB; kk += 32) {
    v16bf a = load_a16(ap + kk, abase);
#pragma unroll
    for (int j = 0; j < 4; ++j) {
      v16bf b = *(const v16bf*)(bp[j] + kk);
      c[j] = wmma_bf16(a, b, c[j]);
    }
  }

  // C layout: VGPR r, lanes 0-15 -> (M=r, N=lane); lanes 16-31 -> (M=r+8).
#pragma unroll
  for (int j = 0; j < 4; ++j) {
    int col   = n0 + j * 16 + (lane & 15);
    int which = col / (NHEADS * HDIM);
    int rem   = col - which * (NHEADS * HDIM);
    int h     = rem >> 6;
    int d     = rem & 63;
    __bf16* dst = (which == 0) ? q : (which == 1) ? k : v;
#pragma unroll
    for (int r = 0; r < 8; ++r) {
      int row = m0 + r + ((lane >= 16) ? 8 : 0);
      int b_  = row >> 10;
      int t   = row & (TLEN - 1);
      dst[(((size_t)b_ * NHEADS + h) * TLEN + t) * HDIM + d] = (__bf16)c[j][r];
    }
  }
}

// ---------------------------------------------------------------------------
// Kernel 4: attention. 4 waves/block; each wave owns a 16-row query tile of
// one (b,h). 64 keys/iteration staged in LDS: K tile via one TDM descriptor
// (fallback: b128 copies), V transposed via coalesced b128 loads + b16 LDS
// scatter, next chunk prefetched. Fixed-max softmax: because logits are
// clamped to [-30,30] before the row max, exp(s-m)<=1 and the 1e5 clamp
// never binds; we accumulate p'=exp(s-30) with no rescale and recover the
// exact reference denominator sum(p)+1e-10 as sum(p') + 1e-10*exp(m-30).
// Query keep-mask ((t%64)%3 != 0) applied on Q load; bf16 output.
// ---------------------------------------------------------------------------
__global__ __launch_bounds__(128) void mha_attn_kernel(
    const __bf16* __restrict__ q, const __bf16* __restrict__ k,
    const __bf16* __restrict__ v, __bf16* __restrict__ attnb) {
  __shared__ __bf16 Kl[64 * 64];     // [key][d]
  __shared__ __bf16 Vt[64 * 64];     // [d][key]
  __shared__ float  Ps[4][16 * 64];  // per-wave P scratch [row][key]

  int tid  = threadIdx.x;
  int lane = tid & 31;
  int wave = tid >> 5;
  int bh   = blockIdx.x >> 4;        // 0..47
  int mb   = blockIdx.x & 15;
  int b_   = bh / NHEADS;
  int h    = bh % NHEADS;
  int m0   = mb * 64 + wave * 16;

  // Load Q tile into A-matrix registers (d 0..31 and 32..63).
  int arow  = lane & 15;
  int tokq  = m0 + arow;
  bool keep = ((tokq & 63) % 3) != 0;
  int abase = (lane < 16) ? 0 : 8;
  const __bf16* qrow = q + ((size_t)bh * TLEN + tokq) * HDIM;
  v16bf aq0 = load_a16(qrow, abase);
  v16bf aq1 = load_a16(qrow + 32, abase);
  if (!keep) { aq0 = zero16bf(); aq1 = zero16bf(); }

  float lmax[8], lsum[8];
#pragma unroll
  for (int r = 0; r < 8; ++r) { lmax[r] = -30.0f; lsum[r] = 0.0f; }
  v8f acc[4];
#pragma unroll
  for (int t = 0; t < 4; ++t) { v8f z = {}; acc[t] = z; }

  const __bf16* kg = k + (size_t)bh * TLEN * HDIM;
  const __bf16* vg = v + (size_t)bh * TLEN * HDIM;
  float* ps = Ps[wave];

  for (int n0 = 0; n0 < TLEN; n0 += 64) {
    __syncthreads();  // previous iteration's LDS reads complete

    // --- Prefetch next chunk (speculative, per-lane 128B lines) ---
    {
      int nn = (n0 + 64 < TLEN) ? (n0 + 64) : n0;
      __builtin_prefetch(kg + (size_t)nn * HDIM + tid * 64, 0, 1);
      __builtin_prefetch(vg + (size_t)nn * HDIM + tid * 64, 0, 1);
    }

    // --- Stage K chunk [64 x 64] bf16 ---
#ifdef MHA_HAS_TDM
    if (wave == 0) {
      tdm_load_tile_64w((unsigned long long)(const void*)(kg + (size_t)n0 * HDIM),
                        lds_byte_offset(Kl), 64);
    }
#else
    {
      const v8bf* ksrc = (const v8bf*)(kg + (size_t)n0 * HDIM);
      v8bf* kdst = (v8bf*)Kl;
      for (int cidx = tid; cidx < 512; cidx += 128) kdst[cidx] = ksrc[cidx];
    }
#endif
    // --- Stage V transposed [d][key]: coalesced row loads, b16 scatter ---
    {
      const __bf16* vsrc = vg + (size_t)n0 * HDIM;
      for (int cidx = tid; cidx < 512; cidx += 128) {
        int key = cidx >> 3;
        int d0  = (cidx & 7) * 8;
        v8bf row = *(const v8bf*)(vsrc + key * HDIM + d0);
#pragma unroll
        for (int i = 0; i < 8; ++i) Vt[(d0 + i) * 64 + key] = row[i];
      }
    }
#ifdef MHA_HAS_TDM
    if (wave == 0) __builtin_amdgcn_s_wait_tensorcnt(0);
#endif
    __syncthreads();

    // Scores: S = Q (16x64) . K^T -> four 16x16 tiles (16 keys each).
    int keyc = lane & 15;
    int doff = (lane < 16) ? 0 : 16;
    v8f s[4];
#pragma unroll
    for (int j = 0; j < 4; ++j) {
      const __bf16* krow = Kl + (j * 16 + keyc) * 64;
      v16bf bk0 = *(const v16bf*)(krow + doff);
      v16bf bk1 = *(const v16bf*)(krow + 32 + doff);
      v8f zz = {};
      s[j] = wmma_bf16(aq1, bk1, wmma_bf16(aq0, bk0, zz));
    }

    // Fixed-max softmax: p' = exp(clamp(s*scale) - 30); no rescale needed.
#pragma unroll
    for (int r = 0; r < 8; ++r) {
      int row = r + ((lane >= 16) ? 8 : 0);
      float psum = 0.0f;
#pragma unroll
      for (int j = 0; j < 4; ++j) {
        float xs = fminf(fmaxf(s[j][r] * 0.125f, -30.0f), 30.0f);
        lmax[r] = fmaxf(lmax[r], xs);
        float p = __expf(xs - 30.0f);
        psum += p;
        ps[row * 64 + j * 16 + keyc] = p;
      }
      lsum[r] += psum;
    }
    __syncthreads();  // P scratch visible wave-wide

    // Re-layout P (16x64) into two A-matrix fragments (keys 0-31, 32-63).
    int prow  = lane & 15;
    int pbase = (lane < 16) ? 0 : 16;
    v16bf pa0, pa1;
#pragma unroll
    for (int i = 0; i < 8; ++i) {
      pa0[i]     = (__bf16)ps[prow * 64 + pbase + i];
      pa0[8 + i] = (__bf16)ps[prow * 64 + pbase + 8 + i];
      pa1[i]     = (__bf16)ps[prow * 64 + 32 + pbase + i];
      pa1[8 + i] = (__bf16)ps[prow * 64 + 32 + pbase + 8 + i];
    }
    // P . V : four 16x16 output tiles across D=64; chained K=32 pairs.
    int dcol = lane & 15;
    int koff = (lane < 16) ? 0 : 16;
#pragma unroll
    for (int t = 0; t < 4; ++t) {
      const __bf16* vrow = Vt + (t * 16 + dcol) * 64;
      v16bf bv0 = *(const v16bf*)(vrow + koff);
      v16bf bv1 = *(const v16bf*)(vrow + 32 + koff);
      acc[t] = wmma_bf16(pa1, bv1, wmma_bf16(pa0, bv0, acc[t]));
    }
  }

  // Epilogue: one cross-lane reduction per row, exact reference denominator:
  // sum(exp(s-m)) + 1e-10  ==  (sum(p') + 1e-10*exp(m-30)) * exp(30-m).
  float inv[8];
#pragma unroll
  for (int r = 0; r < 8; ++r) {
    float m = lmax[r];
    m = fmaxf(m, __shfl_xor(m, 1));
    m = fmaxf(m, __shfl_xor(m, 2));
    m = fmaxf(m, __shfl_xor(m, 4));
    m = fmaxf(m, __shfl_xor(m, 8));
    float l = lsum[r];
    l += __shfl_xor(l, 1);
    l += __shfl_xor(l, 2);
    l += __shfl_xor(l, 4);
    l += __shfl_xor(l, 8);
    inv[r] = 1.0f / (l + 1.0e-10f * __expf(m - 30.0f));
  }
  __bf16* ob = attnb + (size_t)b_ * TLEN * EMB + h * HDIM;
#pragma unroll
  for (int t = 0; t < 4; ++t) {
#pragma unroll
    for (int r = 0; r < 8; ++r) {
      int m   = r + ((lane >= 16) ? 8 : 0);
      int tok = m0 + m;
      ob[(size_t)tok * EMB + t * 16 + (lane & 15)] = (__bf16)(acc[t][r] * inv[r]);
    }
  }
}

// ---------------------------------------------------------------------------
// Kernel 5: output GEMM (16x64 strip per wave) + bias, fp32 out.
// ---------------------------------------------------------------------------
__global__ __launch_bounds__(128) void mha_out_gemm_kernel(
    const __bf16* __restrict__ ab, const __bf16* __restrict__ wo,
    const float* __restrict__ bias, float* __restrict__ out) {
  int lane = threadIdx.x & 31;
  int wave = threadIdx.x >> 5;
  int n0 = blockIdx.x * 64;
  int m0 = (blockIdx.y * 4 + wave) * 16;

  int arow  = m0 + (lane & 15);
  int abase = (lane < 16) ? 0 : 8;
  int bk0   = (lane < 16) ? 0 : 16;

  const __bf16* ap = ab + (size_t)arow * EMB;
  const __bf16* bp[4];
#pragma unroll
  for (int j = 0; j < 4; ++j)
    bp[j] = wo + (size_t)(n0 + j * 16 + (lane & 15)) * EMB + bk0;

  v8f c[4];
#pragma unroll
  for (int j = 0; j < 4; ++j) { v8f z = {}; c[j] = z; }

  for (int kk = 0; kk < EMB; kk += 32) {
    v16bf a = load_a16(ap + kk, abase);
#pragma unroll
    for (int j = 0; j < 4; ++j) {
      v16bf b = *(const v16bf*)(bp[j] + kk);
      c[j] = wmma_bf16(a, b, c[j]);
    }
  }

#pragma unroll
  for (int j = 0; j < 4; ++j) {
    int col = n0 + j * 16 + (lane & 15);
    float bb = bias[col];
#pragma unroll
    for (int r = 0; r < 8; ++r) {
      int row = m0 + r + ((lane >= 16) ? 8 : 0);
      out[(size_t)row * EMB + col] = c[j][r] + bb;
    }
  }
}

// ---------------------------------------------------------------------------
// Host launcher.
// ---------------------------------------------------------------------------
extern "C" void kernel_launch(void* const* d_in, const int* in_sizes, int n_in,
                              void* d_out, int out_size, void* d_ws, size_t ws_size,
                              hipStream_t stream) {
  const float* x    = (const float*)d_in[0];
  const int*   idx  = (const int*)d_in[1];
  const float* se   = (const float*)d_in[2];
  const float* wqkv = (const float*)d_in[3];
  const float* wout = (const float*)d_in[4];
  const float* bout = (const float*)d_in[5];
  float* out = (float*)d_out;

  char* ws = (char*)d_ws;
  __bf16* xb    = (__bf16*)ws; ws += (size_t)NTOK * EMB * 2;
  __bf16* wqb   = (__bf16*)ws; ws += (size_t)QKVN * EMB * 2;
  __bf16* wob   = (__bf16*)ws; ws += (size_t)EMB * EMB * 2;
  __bf16* q     = (__bf16*)ws; ws += (size_t)NTOK * EMB * 2;   // [B,H,T,D]
  __bf16* k     = (__bf16*)ws; ws += (size_t)NTOK * EMB * 2;
  __bf16* v     = (__bf16*)ws; ws += (size_t)NTOK * EMB * 2;
  __bf16* attnb = (__bf16*)ws; ws += (size_t)NTOK * EMB * 2;

  // 1. struct-embed add -> bf16 activations
  {
    int n = NTOK * EMB;
    mha_add_struct_kernel<<<(n + 255) / 256, 256, 0, stream>>>(x, idx, se, xb, n);
  }
  // 2. weight converts
  {
    int n = QKVN * EMB;
    mha_cvt_bf16_kernel<<<(n + 255) / 256, 256, 0, stream>>>(wqkv, wqb, n);
    n = EMB * EMB;
    mha_cvt_bf16_kernel<<<(n + 255) / 256, 256, 0, stream>>>(wout, wob, n);
  }
  // 3. QKV projection (WMMA, 16x64 strip per wave)
  {
    dim3 grid(QKVN / 64, NTOK / 64);   // 36 x 64
    mha_qkv_gemm_kernel<<<grid, 128, 0, stream>>>(xb, wqb, q, k, v);
  }
  // 4. attention (WMMA flash loop, TDM-staged K tiles, prefetch)
  {
    dim3 grid(BATCH * NHEADS * (TLEN / 64));  // 48 * 16 = 768
    mha_attn_kernel<<<grid, 128, 0, stream>>>(q, k, v, attnb);
  }
  // 5. output projection + bias (WMMA)
  {
    dim3 grid(EMB / 64, NTOK / 64);    // 12 x 64
    mha_out_gemm_kernel<<<grid, 128, 0, stream>>>(attnb, wob, bout, out);
  }
}
